// AEVComputer_52158082843350
// MI455X (gfx1250) — matrix operand order repaired and checked by reading, not compiled, not run on hardware
//
#include <hip/hip_runtime.h>

typedef __attribute__((ext_vector_type(16))) _Float16 v16h;
typedef __attribute__((ext_vector_type(8)))  float    v8f;
typedef __attribute__((ext_vector_type(4)))  float    v4f;

#define RCR 5.2f
#define RCA 3.5f
#define PI_F 3.14159265358979323846f

// Compiler-only memory fence: orders this wave's LDS staging stores vs. its own
// gather loads (DS ops from one wave complete in order; no HW barrier needed).
#define WAVE_STAGE_FENCE() asm volatile("" ::: "memory")

__device__ __forceinline__ v8f wmma16x16x32(v16h a, v16h b, v8f c) {
    return __builtin_amdgcn_wmma_f32_16x16x32_f16(
        /*neg_a=*/false, a, /*neg_b=*/false, b,
        /*c_mod=*/(short)0, c, /*reuse_a=*/false, /*reuse_b=*/false);
}

// Gather one 16x32 f16 A-fragment from an LDS tile laid out [m][k] (k contiguous).
// Per ISA layout: lane<16 -> row m, K = {koff..koff+7, koff+16..koff+23} with koff=0;
// lane>=16 -> same row, koff=8.
__device__ __forceinline__ v16h load_A(const _Float16* buf, int m, int koff) {
    v16h a;
    v4f* pa = (v4f*)&a;
    pa[0] = *(const v4f*)(buf + m * 32 + koff);        // K = koff .. koff+7
    pa[1] = *(const v4f*)(buf + m * 32 + koff + 16);   // K = koff+16 .. koff+23
    return a;
}

// Build one-hot B (32x16, f16) from 32 class bytes: B[k][n] = (cls[k]==n).
// Layout: lane = column n (mod 16); lanes 0-15 hold K={0..7,16..23}, lanes 16-31 K={8..15,24..31}.
__device__ __forceinline__ v16h make_B(const unsigned char* cls, int lane) {
    const unsigned int ncol = (unsigned int)(lane & 15);
    const int kbase = (lane < 16) ? 0 : 8;
    unsigned long long lo = *(const unsigned long long*)(cls + kbase);
    unsigned long long hi = *(const unsigned long long*)(cls + kbase + 16);
    v16h b;
#pragma unroll
    for (int e = 0; e < 16; ++e) {
        unsigned int byte =
            (unsigned int)(((e < 8) ? (lo >> (8 * e)) : (hi >> (8 * (e - 8)))) & 0xFFull);
        b[e] = (byte == ncol) ? (_Float16)1.0f : (_Float16)0.0f;
    }
    return b;
}

__global__ __launch_bounds__(256, 2)
void aev_kernel(const int* __restrict__ species,
                const float* __restrict__ coords,
                float* __restrict__ out /* base of aev region */) {
    // ---- LDS ----
    __shared__ float sdist[1024];
    __shared__ float sfcr[1024];   // fc_r * (i!=j) * (d<=RCR)
    __shared__ float sfca[1024];   // fc_a * (i!=j) * (d<=RCA)
    __shared__ float sinvd[1024];
    __shared__ float scx[32], scy[32], scz[32];
    __shared__ __align__(8) unsigned char sspb[32];
    __shared__ unsigned short pairtab[512];                 // 496 used, (j<<8)|k
    __shared__ __align__(16) _Float16 termhi[8][1024];      // per-wave [feat][pair]
    __shared__ __align__(16) _Float16 termlo[8][1024];
    __shared__ __align__(8) unsigned char pcls[8][32];      // per-wave pair classes

    const int n    = blockIdx.x;
    const int tid  = threadIdx.x;
    const int lane = tid & 31;
    const int wave = tid >> 5;
    const int i    = blockIdx.y * 8 + wave;   // center atom

    // ---- phase 0: geometry tables (cross-wave -> real barrier) ----
    if (tid < 32) {
        scx[tid] = coords[(n * 32 + tid) * 3 + 0];
        scy[tid] = coords[(n * 32 + tid) * 3 + 1];
        scz[tid] = coords[(n * 32 + tid) * 3 + 2];
        sspb[tid] = (unsigned char)species[n * 32 + tid];
    }
    __syncthreads();
#pragma unroll
    for (int r = 0; r < 4; ++r) {
        int idx = tid + 256 * r;
        int a = idx >> 5, b = idx & 31;
        float dx = scx[b] - scx[a];
        float dy = scy[b] - scy[a];
        float dz = scz[b] - scz[a];
        float d2 = dx * dx + dy * dy + dz * dz;
        float d = sqrtf(d2 > 0.0f ? d2 : 1.0f);
        sdist[idx] = d;
        sinvd[idx] = 1.0f / d;
        bool diag = (a == b);
        sfcr[idx] = (!diag && d <= RCR) ? (0.5f * __cosf(PI_F * d / RCR) + 0.5f) : 0.0f;
        sfca[idx] = (!diag && d <= RCA) ? (0.5f * __cosf(PI_F * d / RCA) + 0.5f) : 0.0f;
        if (a < b) {
            int p = a * (63 - a) / 2 + b - a - 1;   // 0..495
            pairtab[p] = (unsigned short)((a << 8) | b);
        }
    }
    __syncthreads();   // tables ready for all waves

    const float xi = scx[i], yi = scy[i], zi = scz[i];
    float* op = out + (size_t)(n * 32 + i) * 384;
    const int ncol  = lane & 15;
    const int mbase = (lane < 16) ? 0 : 8;   // C/D row base for this lane
    const int koff  = (lane < 16) ? 0 : 8;   // A K-offset for this lane

    // ---- phase 1: radial (one K=32 WMMA, hi/lo split) ----
    {
        const int j = lane;
        float d = sdist[i * 32 + j];
        float base = 0.25f * sfcr[i * 32 + j];
        _Float16* Whi = termhi[wave];
        _Float16* Wlo = termlo[wave];
#pragma unroll
        for (int m = 0; m < 16; ++m) {
            float shf = 0.9f + 0.26875f * (float)m;
            float e = d - shf;
            float g = base * __expf(-16.0f * e * e);
            _Float16 hi = (_Float16)g;
            Whi[m * 32 + j] = hi;
            Wlo[m * 32 + j] = (_Float16)(g - (float)hi);
        }
        WAVE_STAGE_FENCE();   // wave-private staging: no block barrier needed
        v16h B = make_B(sspb, lane);
        v8f acc = {};
        acc = wmma16x16x32(load_A(Whi, lane & 15, koff), B, acc);
        acc = wmma16x16x32(load_A(Wlo, lane & 15, koff), B, acc);
        if (ncol < 4) {
            v4f* dst = (v4f*)(op + ncol * 16 + mbase);
            dst[0] = ((v4f*)&acc)[0];
            dst[1] = ((v4f*)&acc)[1];
        }
        WAVE_STAGE_FENCE();   // order gathers vs. upcoming angular overwrite
    }

    // ---- phase 2: angular (16 chunks x 32 pairs, 2 M-tiles, hi/lo split) ----
    const float CZ[8] = { 0.98078528f,  0.83146961f,  0.55557023f,  0.19509032f,
                         -0.19509032f, -0.55557023f, -0.83146961f, -0.98078528f };
    const float SZ[8] = { 0.19509032f,  0.55557023f,  0.83146961f,  0.98078528f,
                          0.98078528f,  0.83146961f,  0.55557023f,  0.19509032f };
    const float SHFA[4] = { 0.9f, 1.55f, 2.2f, 2.85f };

    v8f acc0 = {};   // features 0..15
    v8f acc1 = {};   // features 16..31
    _Float16* Whi = termhi[wave];
    _Float16* Wlo = termlo[wave];

    for (int c = 0; c < 16; ++c) {
        int t = c * 32 + lane;
        bool valid = t < 496;
        unsigned short jk = pairtab[valid ? t : 0];
        int j = jk >> 8, k = jk & 255;

        float dij = sdist[i * 32 + j];
        float dik = sdist[i * 32 + k];
        float fc2 = valid ? (2.0f * sfca[i * 32 + j] * sfca[i * 32 + k]) : 0.0f;

        float dxj = scx[j] - xi, dyj = scy[j] - yi, dzj = scz[j] - zi;
        float dxk = scx[k] - xi, dyk = scy[k] - yi, dzk = scz[k] - zi;
        float dot = dxj * dxk + dyj * dyk + dzj * dzk;
        float cosang = 0.95f * dot * sinvd[i * 32 + j] * sinvd[i * 32 + k];
        cosang = fminf(0.95f, fmaxf(-0.95f, cosang));
        float sinang = sqrtf(1.0f - cosang * cosang);
        float dm = 0.5f * (dij + dik);

        float f2v[4];
#pragma unroll
        for (int a = 0; a < 4; ++a) {
            float e = dm - SHFA[a];
            f2v[a] = __expf(-8.0f * e * e);
        }
        float f1v[8];
#pragma unroll
        for (int z = 0; z < 8; ++z) {
            float b0 = 0.5f + 0.5f * (cosang * CZ[z] + sinang * SZ[z]);
            float p = b0 * b0;  // ^2
            p *= p;             // ^4
            p *= p;             // ^8
            p *= p;             // ^16
            p *= p;             // ^32
            f1v[z] = p;
        }
#pragma unroll
        for (int a = 0; a < 4; ++a) {
            float fa = fc2 * f2v[a];
#pragma unroll
            for (int z = 0; z < 8; ++z) {
                int u = a * 8 + z;
                float v = fa * f1v[z];
                _Float16 hi = (_Float16)v;
                Whi[u * 32 + lane] = hi;
                Wlo[u * 32 + lane] = (_Float16)(v - (float)hi);
            }
        }
        {   // pair class (min/max species -> triu index)
            int sa = sspb[j], sb = sspb[k];
            int lo = min(sa, sb), hi = max(sa, sb);
            pcls[wave][lane] = (unsigned char)(lo * 4 - (lo * (lo + 1)) / 2 + hi);
        }
        WAVE_STAGE_FENCE();   // stores -> gathers (same wave, DS in-order)

        v16h B = make_B(pcls[wave], lane);
        acc0 = wmma16x16x32(load_A(Whi, (lane & 15),      koff), B, acc0);
        acc0 = wmma16x16x32(load_A(Wlo, (lane & 15),      koff), B, acc0);
        acc1 = wmma16x16x32(load_A(Whi, (lane & 15) + 16, koff), B, acc1);
        acc1 = wmma16x16x32(load_A(Wlo, (lane & 15) + 16, koff), B, acc1);
        WAVE_STAGE_FENCE();   // gathers -> next chunk's stores
    }

    // ---- write angular: out idx = 64 + p*32 + u ----
    if (ncol < 10) {
        v4f* d0 = (v4f*)(op + 64 + ncol * 32 + mbase);
        d0[0] = ((v4f*)&acc0)[0];
        d0[1] = ((v4f*)&acc0)[1];
        v4f* d1 = (v4f*)(op + 64 + ncol * 32 + 16 + mbase);
        d1[0] = ((v4f*)&acc1)[0];
        d1[1] = ((v4f*)&acc1)[1];
    }
}

__global__ void copy_species_kernel(const int* __restrict__ sp, int* __restrict__ out) {
    int t = blockIdx.x * 256 + threadIdx.x;
    if (t < 2048) out[t] = sp[t];
}

extern "C" void kernel_launch(void* const* d_in, const int* in_sizes, int n_in,
                              void* d_out, int out_size, void* d_ws, size_t ws_size,
                              hipStream_t stream) {
    (void)in_sizes; (void)n_in; (void)out_size; (void)d_ws; (void)ws_size;
    const int*   species = (const int*)d_in[0];
    const float* coords  = (const float*)d_in[1];

    // d_out = [species (2048 x int32 bits)][aev (64*32*384 f32)]
    copy_species_kernel<<<8, 256, 0, stream>>>(species, (int*)d_out);
    float* aev = (float*)d_out + 2048;
    aev_kernel<<<dim3(64, 4), 256, 0, stream>>>(species, coords, aev);
}
